// ProperTerrainClassifier_11905649345068
// MI455X (gfx1250) — matrix operand order repaired and checked by reading, not compiled, not run on hardware
//
#include <hip/hip_runtime.h>

// ---------------------------------------------------------------------------
// ResNet-50 forward (terrain classifier) for gfx1250 (MI455X).
// Convs as implicit GEMM over NHWC bf16 activations using
// v_wmma_f32_16x16x32_bf16 (wave32).  Per-wave C-tile: 32 (cout) x 64 (pixel)
// = 8 WMMA fragments; K-loop unrolled x2 so the scheduler can overlap the
// next step's 12 b128 loads with the current step's 8 WMMAs (covers the
// bf16 WMMA RAW window from the CDNA5 hazard table).
// BN folded into bf16 weights; bias+residual+ReLU fused in epilogue.
// ---------------------------------------------------------------------------

typedef __attribute__((ext_vector_type(16))) __bf16 v16bf;
typedef __attribute__((ext_vector_type(8)))  float  v8f;

#define BN_EPS 1e-5f

__device__ __forceinline__ unsigned short f2bf(float f) {
  unsigned u = __builtin_bit_cast(unsigned, f);
  u += 0x7FFFu + ((u >> 16) & 1u);            // round-to-nearest-even
  return (unsigned short)(u >> 16);
}
__device__ __forceinline__ float bf2f(unsigned short h) {
  unsigned u = ((unsigned)h) << 16;
  return __builtin_bit_cast(float, u);
}

union Frag16 { uint4 q[2]; v16bf v; };

// ---------------------------------------------------------------------------
// Weight repack: OIHW f32 -> [kpos][Cout][Cin] bf16, BN scale folded in.
// ---------------------------------------------------------------------------
__global__ void k_pack_w(const float* __restrict__ w,
                         const float* __restrict__ g,
                         const float* __restrict__ v,
                         unsigned short* __restrict__ wout,
                         int Cout, int Cin, int ks) {
  long total = (long)Cout * Cin * ks * ks;
  long gid = (long)blockIdx.x * blockDim.x + threadIdx.x;
  if (gid >= total) return;
  long cc = (long)Cout * Cin;
  int kpos = (int)(gid / cc);
  int r    = (int)(gid % cc);
  int co = r / Cin, ci = r % Cin;
  int kh = kpos / ks, kw = kpos % ks;
  float sc = g[co] * rsqrtf(v[co] + BN_EPS);
  float val = w[(((long)co * Cin + ci) * ks + kh) * ks + kw] * sc;
  wout[gid] = f2bf(val);
}

__global__ void k_bias(const float* __restrict__ g, const float* __restrict__ b,
                       const float* __restrict__ m, const float* __restrict__ v,
                       float* __restrict__ bout, int C) {
  int i = blockIdx.x * blockDim.x + threadIdx.x;
  if (i >= C) return;
  float sc = g[i] * rsqrtf(v[i] + BN_EPS);
  bout[i] = b[i] - m[i] * sc;
}

// ---------------------------------------------------------------------------
// Stem: 7x7 s2 p3 conv (Cin=3) + BN + ReLU.  f32 NCHW in -> NHWC bf16 out.
// ---------------------------------------------------------------------------
__global__ void k_conv1(const float* __restrict__ x, const float* __restrict__ w,
                        const float* __restrict__ g, const float* __restrict__ b,
                        const float* __restrict__ m, const float* __restrict__ v,
                        unsigned short* __restrict__ out) {
  const long total = 32L * 112 * 112 * 64;
  long gid = (long)blockIdx.x * blockDim.x + threadIdx.x;
  if (gid >= total) return;
  int co = (int)(gid & 63);
  long pl = gid >> 6;
  int n = (int)(pl / (112 * 112));
  int rem = (int)(pl % (112 * 112));
  int ho = rem / 112, wo = rem % 112;
  float acc = 0.f;
  for (int c = 0; c < 3; ++c) {
    const float* xp = x + ((long)n * 3 + c) * 224 * 224;
    const float* wp = w + ((long)co * 3 + c) * 49;
    for (int kh = 0; kh < 7; ++kh) {
      int hin = ho * 2 - 3 + kh;
      if (hin < 0 || hin >= 224) continue;
      for (int kw = 0; kw < 7; ++kw) {
        int win = wo * 2 - 3 + kw;
        if (win < 0 || win >= 224) continue;
        acc += xp[hin * 224 + win] * wp[kh * 7 + kw];
      }
    }
  }
  float sc = g[co] * rsqrtf(v[co] + BN_EPS);
  acc = fmaxf(acc * sc + (b[co] - m[co] * sc), 0.f);
  out[gid] = f2bf(acc);
}

// 3x3 s2 p1 maxpool on NHWC bf16 (112x112x64 -> 56x56x64)
__global__ void k_maxpool(const unsigned short* __restrict__ in,
                          unsigned short* __restrict__ out) {
  const long total = 32L * 56 * 56 * 64;
  long gid = (long)blockIdx.x * blockDim.x + threadIdx.x;
  if (gid >= total) return;
  int c = (int)(gid & 63);
  long pl = gid >> 6;
  int n = (int)(pl / (56 * 56));
  int rem = (int)(pl % (56 * 56));
  int ho = rem / 56, wo = rem % 56;
  float mx = -3.4e38f;
  for (int kh = 0; kh < 3; ++kh) {
    int hin = ho * 2 - 1 + kh;
    if (hin < 0 || hin >= 112) continue;
    for (int kw = 0; kw < 3; ++kw) {
      int win = wo * 2 - 1 + kw;
      if (win < 0 || win >= 112) continue;
      mx = fmaxf(mx, bf2f(in[(((long)n * 112 + hin) * 112 + win) * 64 + c]));
    }
  }
  out[gid] = f2bf(mx);
}

// ---------------------------------------------------------------------------
// Implicit-GEMM conv via WMMA bf16. One wave per block, 32x64 C-tile:
//   A = weights [kpos][Cout][Cin] bf16   (M = cout, K = cin per kpos)
//   B = activations NHWC bf16           (N = pixel, K = cin contiguous)
// 2 A-frags x 4 B-frags -> 8 WMMAs per 32-deep K-step (A reused 4x).
// Epilogue: bias + optional bf16 residual + optional ReLU -> NHWC bf16.
// EXEC is full at every WMMA (blockDim=32, uniform control flow there).
// ---------------------------------------------------------------------------
__global__ __launch_bounds__(32) void k_conv_wmma(
    const unsigned short* __restrict__ in, const unsigned short* __restrict__ wgt,
    const float* __restrict__ bias, const unsigned short* __restrict__ resid,
    unsigned short* __restrict__ out,
    int Hin, int Win, int Cin, int Hout, int Wout, int Cout,
    int ks, int stride, int pad, int relu) {
  const int lane = threadIdx.x;
  const int l15 = lane & 15;
  const int hi  = lane >> 4;
  const int HWo = Hout * Wout;
  const int totalP = 32 * HWo;
  const int nBase = blockIdx.x * 64;
  const int mBase = blockIdx.y * 32;

  int pix[4], pn[4], ph[4], pw[4];
  bool pv[4];
#pragma unroll
  for (int j = 0; j < 4; ++j) {
    int p = nBase + j * 16 + l15;
    pv[j] = p < totalP;
    int pp = pv[j] ? p : 0;
    pn[j] = pp / HWo;
    int rem = pp % HWo;
    ph[j] = rem / Wout;
    pw[j] = rem % Wout;
    pix[j] = p;
  }

  v8f c0[4] = {v8f{}, v8f{}, v8f{}, v8f{}};   // A-frag 0 (couts mBase..+15)
  v8f c1[4] = {v8f{}, v8f{}, v8f{}, v8f{}};   // A-frag 1 (couts mBase+16..+31)

  // Per-lane A row offsets (constant across K loop)
  const long aRow0 = ((long)(mBase + l15)) * Cin + hi * 8;
  const long aRow1 = ((long)(mBase + 16 + l15)) * Cin + hi * 8;

  const int kk = ks * ks;
  for (int kpos = 0; kpos < kk; ++kpos) {
    int kh = kpos / ks, kw = kpos % ks;
    long base[4];
#pragma unroll
    for (int j = 0; j < 4; ++j) {
      int hin = ph[j] * stride - pad + kh;
      int win = pw[j] * stride - pad + kw;
      bool ok = pv[j] && hin >= 0 && hin < Hin && win >= 0 && win < Win;
      base[j] = ok ? (((long)pn[j] * Hin + hin) * Win + win) * (long)Cin + hi * 8
                   : -1L;
    }
    const unsigned short* wk = wgt + (long)kpos * Cout * Cin;
    // Cin is always a multiple of 64 here, so unroll-by-2 has no remainder;
    // two independent K-steps in flight let loads of step k+1 overlap the
    // WMMAs of step k and hide the WMMA RAW window.
#pragma unroll 2
    for (int k0 = 0; k0 < Cin; k0 += 32) {
      Frag16 a0, a1, b[4];
      // A layout: lane m holds K {0..7,16..23} (lanes 0-15) / {8..15,24..31}
      {
        const unsigned short* ap = wk + aRow0 + k0;
        a0.q[0] = *(const uint4*)ap;
        a0.q[1] = *(const uint4*)(ap + 16);
      }
      {
        const unsigned short* ap = wk + aRow1 + k0;
        a1.q[0] = *(const uint4*)ap;
        a1.q[1] = *(const uint4*)(ap + 16);
      }
      // B layout: lane n holds pixel column, K {0..7,16..23} (lanes 0-15) / {8..15,24..31}
#pragma unroll
      for (int j = 0; j < 4; ++j) {
        if (base[j] >= 0) {
          const unsigned short* bp = in + base[j] + k0;
          b[j].q[0] = *(const uint4*)bp;
          b[j].q[1] = *(const uint4*)(bp + 16);
        } else {
          b[j].q[0] = make_uint4(0, 0, 0, 0);
          b[j].q[1] = make_uint4(0, 0, 0, 0);
        }
      }
#pragma unroll
      for (int j = 0; j < 4; ++j) {
        c0[j] = __builtin_amdgcn_wmma_f32_16x16x32_bf16(false, a0.v, false, b[j].v,
                                                        (short)0, c0[j], false, false);
        c1[j] = __builtin_amdgcn_wmma_f32_16x16x32_bf16(false, a1.v, false, b[j].v,
                                                        (short)0, c1[j], false, false);
      }
    }
  }

  // Epilogue: C layout — VGPR r: M = r (lanes 0-15) / r+8 (lanes 16-31), N = lane&15
#pragma unroll
  for (int i = 0; i < 2; ++i) {
#pragma unroll
    for (int j = 0; j < 4; ++j) {
      if (!pv[j]) continue;
      v8f cf = (i == 0) ? c0[j] : c1[j];
      int co0 = mBase + i * 16 + hi * 8;
      long oo = (long)pix[j] * Cout + co0;
      float vv[8];
#pragma unroll
      for (int r = 0; r < 8; ++r) vv[r] = cf[r] + bias[co0 + r];
      if (resid) {
        uint4 rq = *(const uint4*)(resid + oo);
        unsigned rr[4] = {rq.x, rq.y, rq.z, rq.w};
#pragma unroll
        for (int t = 0; t < 4; ++t) {
          vv[2 * t]     += bf2f((unsigned short)(rr[t] & 0xffffu));
          vv[2 * t + 1] += bf2f((unsigned short)(rr[t] >> 16));
        }
      }
      if (relu) {
#pragma unroll
        for (int r = 0; r < 8; ++r) vv[r] = fmaxf(vv[r], 0.f);
      }
      uint4 q;
      q.x = (unsigned)f2bf(vv[0]) | ((unsigned)f2bf(vv[1]) << 16);
      q.y = (unsigned)f2bf(vv[2]) | ((unsigned)f2bf(vv[3]) << 16);
      q.z = (unsigned)f2bf(vv[4]) | ((unsigned)f2bf(vv[5]) << 16);
      q.w = (unsigned)f2bf(vv[6]) | ((unsigned)f2bf(vv[7]) << 16);
      *(uint4*)(out + oo) = q;
    }
  }
}

// ---------------------------------------------------------------------------
// Head
// ---------------------------------------------------------------------------
__global__ void k_avgpool(const unsigned short* __restrict__ in,
                          float* __restrict__ feat) {
  int gid = blockIdx.x * blockDim.x + threadIdx.x;
  if (gid >= 32 * 2048) return;
  int n = gid >> 11, c = gid & 2047;
  float s = 0.f;
  for (int h = 0; h < 7; ++h)
    for (int w = 0; w < 7; ++w)
      s += bf2f(in[(((long)n * 7 + h) * 7 + w) * 2048 + c]);
  feat[gid] = s / 49.f;
}

__global__ void k_fc(const float* __restrict__ a, const float* __restrict__ w,
                     const float* __restrict__ b, float* __restrict__ out,
                     int B, int O, int K, int relu) {
  int gid = blockIdx.x * blockDim.x + threadIdx.x;
  if (gid >= B * O) return;
  int n = gid / O, o = gid % O;
  const float* ar = a + (long)n * K;
  const float* wr = w + (long)o * K;
  float acc = b[o];
  for (int k = 0; k < K; ++k) acc += ar[k] * wr[k];
  if (relu) acc = fmaxf(acc, 0.f);
  out[gid] = acc;
}

// Per-(n,c) mean over 224x224 of f32 NCHW input x -> mr[32][3]
__global__ void k_chan_mean(const float* __restrict__ x, float* __restrict__ mr) {
  int n = blockIdx.x / 3, c = blockIdx.x % 3;
  const float* p = x + ((long)n * 3 + c) * 50176;
  float s = 0.f;
  for (int i = threadIdx.x; i < 50176; i += 256) s += p[i];
  __shared__ float sm[256];
  sm[threadIdx.x] = s;
  __syncthreads();
  for (int st = 128; st > 0; st >>= 1) {
    if (threadIdx.x < st) sm[threadIdx.x] += sm[threadIdx.x + st];
    __syncthreads();
  }
  if (threadIdx.x == 0) mr[blockIdx.x] = sm[0] / 50176.f;
}

__global__ void k_final(const float* __restrict__ logits,
                        const float* __restrict__ mr, float* __restrict__ out) {
  int n = threadIdx.x;
  if (n >= 32) return;
  float t[6];
  float mx = -3.4e38f;
  for (int i = 0; i < 6; ++i) { t[i] = logits[n * 6 + i]; mx = fmaxf(mx, t[i]); }
  float se = 0.f;
  for (int i = 0; i < 6; ++i) { t[i] = __expf(t[i] - mx); se += t[i]; }
  for (int i = 0; i < 6; ++i) t[i] /= se;

  float r = mr[n * 3 + 0], g = mr[n * 3 + 1], b = mr[n * 3 + 2];
  float mean = (r + g + b) * (1.f / 3.f);
  float var = ((r - mean) * (r - mean) + (g - mean) * (g - mean) +
               (b - mean) * (b - mean)) * (1.f / 3.f);
  float col[6];
  col[0] = 0.2f * (var < 0.05f);
  col[1] = 0.3f * ((b > r) && (b > g));
  col[2] = 0.3f * ((g > r) && (g > b));
  col[3] = 0.2f * ((g > 0.3f) && (g < 0.7f) && (g > b));
  col[4] = 0.2f * ((r > 0.3f) && (r < 0.7f) && (g > 0.3f) && (g < 0.7f) &&
                   (b > 0.3f) && (b < 0.7f));
  col[5] = 0.2f * (mean > 0.6f);

  float cb[6];
  mx = -3.4e38f;
  for (int i = 0; i < 6; ++i) { cb[i] = 0.8f * t[i] + 0.2f * col[i]; mx = fmaxf(mx, cb[i]); }
  se = 0.f;
  float e[6];
  for (int i = 0; i < 6; ++i) { e[i] = __expf(cb[i] - mx); se += e[i]; }
  for (int i = 0; i < 6; ++i) out[n * 6 + i] = e[i] / se;
}

// ---------------------------------------------------------------------------
// Host orchestration (graph-capture safe: kernel launches only)
// ---------------------------------------------------------------------------
extern "C" void kernel_launch(void* const* d_in, const int* in_sizes, int n_in,
                              void* d_out, int out_size, void* d_ws, size_t ws_size,
                              hipStream_t stream) {
  (void)in_sizes; (void)n_in; (void)out_size; (void)ws_size;
  int ii = 0;
  auto F = [&](void) -> const float* { return (const float*)d_in[ii++]; };

  const float* x = F();
  const float* c1w = F();
  const float* bn1g = F(); const float* bn1b = F();
  const float* bn1m = F(); const float* bn1v = F();

  struct Bn { const float *g, *b, *m, *v; };
  struct Blk {
    const float* w1; Bn b1; const float* w2; Bn b2; const float* w3; Bn b3;
    const float* dw; Bn db; int hasDw, width, stride, inC;
  };
  Blk blk[16];
  const int sw[4] = {64, 128, 256, 512};
  const int sn[4] = {3, 4, 6, 3};
  const int ss[4] = {1, 2, 2, 2};
  int bi = 0, inC = 64;
  for (int s = 0; s < 4; ++s) {
    for (int k = 0; k < sn[s]; ++k) {
      Blk& B = blk[bi];
      B.w1 = F(); B.b1 = {F(), F(), F(), F()};
      B.w2 = F(); B.b2 = {F(), F(), F(), F()};
      B.w3 = F(); B.b3 = {F(), F(), F(), F()};
      B.hasDw = (k == 0);
      if (B.hasDw) { B.dw = F(); B.db = {F(), F(), F(), F()}; } else { B.dw = nullptr; }
      B.width = sw[s];
      B.stride = (k == 0) ? ss[s] : 1;
      B.inC = inC;
      inC = 4 * sw[s];
      ++bi;
    }
  }
  const float* hW1 = F(); const float* hb1 = F();
  const float* hW2 = F(); const float* hb2 = F();
  const float* hW3 = F(); const float* hb3 = F();

  // ---- workspace carve (bf16 activations, bf16 weight arena, f32 scratch) --
  const long BIG = 25690112L;   // 32*112*112*64 elems
  const long MID = 12845056L;   // 32*56*56*128 elems
  const long WRE = 25165824L;   // bf16 weight arena (needs ~23.45M)
  char* p = (char*)d_ws;
  unsigned short* bufA = (unsigned short*)p; p += BIG * 2;
  unsigned short* bufB = (unsigned short*)p; p += BIG * 2;
  unsigned short* bufC = (unsigned short*)p; p += BIG * 2;
  unsigned short* mid1 = (unsigned short*)p; p += MID * 2;
  unsigned short* mid2 = (unsigned short*)p; p += MID * 2;
  unsigned short* wreg = (unsigned short*)p; p += WRE * 2;
  float* breg   = (float*)p; p += 32768L * 4;
  float* feat   = (float*)p; p += 65536L * 4;
  float* z1     = (float*)p; p += 16384L * 4;
  float* z2     = (float*)p; p += 8192L * 4;
  float* logits = (float*)p; p += 256L * 4;
  float* mr     = (float*)p; p += 128L * 4;

  // ---- weight prep: fold BN, repack to [kpos][Cout][Cin] bf16 --------------
  long woff = 0, boff = 0;
  long ow1[16], ow2[16], ow3[16], owd[16] = {0};
  long ob1[16], ob2[16], ob3[16], obd[16] = {0};
  auto pack = [&](const float* w, Bn bn, int Cout, int Cin, int ks,
                  long& wo, long& bo) {
    wo = woff; bo = boff;
    long ne = (long)Cout * Cin * ks * ks;
    k_pack_w<<<dim3((unsigned)((ne + 255) / 256)), dim3(256), 0, stream>>>(
        w, bn.g, bn.v, wreg + woff, Cout, Cin, ks);
    k_bias<<<dim3((unsigned)((Cout + 255) / 256)), dim3(256), 0, stream>>>(
        bn.g, bn.b, bn.m, bn.v, breg + boff, Cout);
    woff += ne; boff += Cout;
  };
  for (int b = 0; b < 16; ++b) {
    Blk& B = blk[b];
    pack(B.w1, B.b1, B.width, B.inC, 1, ow1[b], ob1[b]);
    pack(B.w2, B.b2, B.width, B.width, 3, ow2[b], ob2[b]);
    pack(B.w3, B.b3, 4 * B.width, B.width, 1, ow3[b], ob3[b]);
    if (B.hasDw) pack(B.dw, B.db, 4 * B.width, B.inC, 1, owd[b], obd[b]);
  }

  // ---- stem ----------------------------------------------------------------
  {
    long tot = 32L * 112 * 112 * 64;
    k_conv1<<<dim3((unsigned)((tot + 255) / 256)), 256, 0, stream>>>(
        x, c1w, bn1g, bn1b, bn1m, bn1v, bufB);
    long tp = 32L * 56 * 56 * 64;
    k_maxpool<<<dim3((unsigned)((tp + 255) / 256)), 256, 0, stream>>>(bufB, bufA);
  }

  auto conv = [&](const unsigned short* in_, long wo, long bo,
                  const unsigned short* rs, unsigned short* outp,
                  int Hin, int Win, int Cin_, int Ho, int Wo, int Cout,
                  int ks, int st, int pd, int relu) {
    long totalP = 32L * Ho * Wo;
    dim3 grid((unsigned)((totalP + 63) / 64), (unsigned)(Cout / 32));
    k_conv_wmma<<<grid, 32, 0, stream>>>(in_, wreg + wo, breg + bo, rs, outp,
                                         Hin, Win, Cin_, Ho, Wo, Cout, ks, st,
                                         pd, relu);
  };

  // ---- bottleneck stages ---------------------------------------------------
  int H = 56, W = 56;
  inC = 64;
  for (int b = 0; b < 16; ++b) {
    Blk& B = blk[b];
    int s = B.stride;
    int Ho = (s == 2) ? H / 2 : H;
    int Wo = (s == 2) ? W / 2 : W;
    conv(bufA, ow1[b], ob1[b], nullptr, mid1, H, W, inC, H, W, B.width, 1, 1, 0, 1);
    conv(mid1, ow2[b], ob2[b], nullptr, mid2, H, W, B.width, Ho, Wo, B.width, 3, s, 1, 1);
    const unsigned short* rs;
    if (B.hasDw) {
      conv(bufA, owd[b], obd[b], nullptr, bufC, H, W, inC, Ho, Wo, 4 * B.width, 1, s, 0, 0);
      rs = bufC;
    } else {
      rs = bufA;
    }
    conv(mid2, ow3[b], ob3[b], rs, bufB, Ho, Wo, B.width, Ho, Wo, 4 * B.width, 1, 1, 0, 1);
    unsigned short* t = bufA; bufA = bufB; bufB = t;
    H = Ho; W = Wo; inC = 4 * B.width;
  }

  // ---- head ----------------------------------------------------------------
  k_avgpool<<<dim3((32 * 2048 + 255) / 256), 256, 0, stream>>>(bufA, feat);
  k_fc<<<dim3((32 * 512 + 255) / 256), 256, 0, stream>>>(feat, hW1, hb1, z1, 32, 512, 2048, 1);
  k_fc<<<dim3((32 * 256 + 255) / 256), 256, 0, stream>>>(z1, hW2, hb2, z2, 32, 256, 512, 1);
  k_fc<<<dim3(1), 256, 0, stream>>>(z2, hW3, hb3, logits, 32, 6, 256, 0);
  k_chan_mean<<<dim3(96), 256, 0, stream>>>(x, mr);
  k_final<<<dim3(1), 32, 0, stream>>>(logits, mr, (float*)d_out);
}
